// HetNet_8151847927963
// MI455X (gfx1250) — compile-verified, matrix operand order checked by reading, not run on hardware
//
#include <hip/hip_runtime.h>
#include <hip/hip_bf16.h>

typedef __attribute__((ext_vector_type(16))) _Float16 v16h;
typedef __attribute__((ext_vector_type(8)))  float    v8f;
typedef int vi4 __attribute__((vector_size(16)));

#define N_NODES 20000
#define N_EDGES 40000
#define SE 46
#define TE 82
#define DD 128          // D
#define G4 512          // 4*D
#define FF 256          // F
#define HID 128
#define XC 384          // F + D

// ---------------------------------------------------------------------------
// CDNA5 async global->LDS staging (ASYNCcnt path), with safe fallback.
// Builtin signature (probed via compile error): param0 = v4i AS1*, param1 =
// v4i AS3*, then imm offset + cpol.
// ---------------------------------------------------------------------------
#if __has_builtin(__builtin_amdgcn_global_load_async_to_lds_b128)
#define HAVE_ASYNC_LDS 1
#else
#define HAVE_ASYNC_LDS 0
#endif

#if __has_builtin(__builtin_amdgcn_s_wait_asynccnt)
#define ASYNC_WAIT() __builtin_amdgcn_s_wait_asynccnt(0)
#else
#define ASYNC_WAIT() asm volatile("s_wait_asynccnt 0x0" ::: "memory")
#endif

typedef __attribute__((address_space(1))) vi4* gvec4p;
typedef __attribute__((address_space(3))) vi4* lvec4p;

// Stage `bytes` (multiple of 16, both pointers 16B aligned) from global into
// LDS, cooperatively across `nthreads` threads.
__device__ __forceinline__ void stage_to_lds(const void* __restrict__ g,
                                             void* lds, int bytes,
                                             int tid, int nthreads) {
#if HAVE_ASYNC_LDS
  for (int off = tid * 16; off < bytes; off += nthreads * 16) {
    __builtin_amdgcn_global_load_async_to_lds_b128(
        (gvec4p)((const char*)g + off),
        (lvec4p)((char*)lds + off),
        0, 0);
  }
  ASYNC_WAIT();
#else
  for (int off = tid * 16; off < bytes; off += nthreads * 16) {
    *(int4*)((char*)lds + off) = *(const int4*)((const char*)g + off);
  }
#endif
}

// ---------------------------------------------------------------------------
// 1) Per-edge scores: h * r * t with sinusoidal time features.
// ---------------------------------------------------------------------------
__global__ void k_edge_scores(const int* __restrict__ heads,
                              const int* __restrict__ rels,
                              const int* __restrict__ tails,
                              const float* __restrict__ weeks,
                              const float* __restrict__ ent,
                              const float* __restrict__ rel_embs,
                              const float* __restrict__ w_freq,
                              const float* __restrict__ w_phi,
                              const float* __restrict__ w_amp,
                              _Float16* __restrict__ scores) {
  int idx = blockIdx.x * blockDim.x + threadIdx.x;
  if (idx >= N_EDGES * DD) return;
  int e = idx >> 7;
  int d = idx & 127;
  int hd = heads[e], tl = tails[e], rl = rels[e];
  float wk = weeks[e];
  float hv, tv;
  if (d < SE) {
    hv = ent[(size_t)hd * SE + d];
    tv = ent[(size_t)tl * SE + d];
  } else {
    int j = d - SE;
    size_t hi = (size_t)hd * TE + j;
    size_t ti = (size_t)tl * TE + j;
    hv = w_amp[hi] * sinf(w_freq[hi] * wk + w_phi[hi]);
    tv = w_amp[ti] * sinf(w_freq[ti] * wk + w_phi[ti]);
  }
  float rv = rel_embs[rl * DD + d];
  scores[idx] = (_Float16)(hv * rv * tv);
}

// ---------------------------------------------------------------------------
// Utility kernels
// ---------------------------------------------------------------------------
__global__ void k_cvt_f16(const float* __restrict__ src, _Float16* __restrict__ dst, int n) {
  int i = blockIdx.x * blockDim.x + threadIdx.x;
  if (i < n) dst[i] = (_Float16)src[i];
}
__global__ void k_zero_f32(float* __restrict__ p, int n) {
  int i = blockIdx.x * blockDim.x + threadIdx.x;
  if (i < n) p[i] = 0.0f;
}
__global__ void k_zero_i32(int* __restrict__ p, int n) {
  int i = blockIdx.x * blockDim.x + threadIdx.x;
  if (i < n) p[i] = 0;
}
__global__ void k_hist(const int* __restrict__ keys, int* __restrict__ cnt, int n) {
  int i = blockIdx.x * blockDim.x + threadIdx.x;
  if (i < n) atomicAdd(&cnt[keys[i]], 1);
}

// Deterministic stable counting-sort placement + prefix scans (single thread;
// tiny work, stable within a bin to match jnp.argsort).  edge_src is
// pre-sorted -> identity permutation.
__global__ void k_sort_build(const int* __restrict__ edge_dst,
                             const int* __restrict__ cnt_src,
                             const int* __restrict__ cnt_dst,
                             int* __restrict__ start_src,
                             int* __restrict__ start_dst,
                             int* __restrict__ perm_s,
                             int* __restrict__ perm_d,
                             int* __restrict__ fill) {
  if (threadIdx.x != 0 || blockIdx.x != 0) return;
  int s = 0;
  for (int n = 0; n < N_NODES; ++n) { start_src[n] = s; s += cnt_src[n]; }
  s = 0;
  for (int n = 0; n < N_NODES; ++n) { start_dst[n] = s; s += cnt_dst[n]; fill[n] = 0; }
  for (int e = 0; e < N_EDGES; ++e) perm_s[e] = e;
  for (int e = 0; e < N_EDGES; ++e) {
    int d = edge_dst[e];
    perm_d[start_dst[d] + fill[d]] = e;
    fill[d]++;
  }
}

// ---------------------------------------------------------------------------
// 2) pre = scores @ Wih^T + bih   [E,128] x [128,512] -> [E,512]
//    Wih staged once per block into LDS (async copy); 8 waves share it.
// ---------------------------------------------------------------------------
__global__ __launch_bounds__(256) void k_pre_gemm(const _Float16* __restrict__ scores,
                                                  const _Float16* __restrict__ Wih,
                                                  const float* __restrict__ bih,
                                                  float* __restrict__ pre) {
  __shared__ _Float16 wih_s[G4 * DD];      // 128 KB

  stage_to_lds(Wih, wih_s, G4 * DD * 2, threadIdx.x, blockDim.x);
  __syncthreads();

  int gwave = (blockIdx.x * blockDim.x + threadIdx.x) >> 5;
  int lane  = threadIdx.x & 31;
  if (gwave >= N_EDGES / 16) return;       // wave-uniform guard (after barrier)
  int row0  = gwave * 16;
  int mloc  = lane & 15;
  int khalf = lane >> 4;

  v16h A[4];
  for (int kc = 0; kc < 4; ++kc)
    A[kc] = *(const v16h*)(scores + (size_t)(row0 + mloc) * DD + kc * 32 + khalf * 16);

  for (int nt = 0; nt < 32; ++nt) {
    v8f acc = {};
    for (int kc = 0; kc < 4; ++kc) {
      v16h B = *(const v16h*)(wih_s + (nt * 16 + mloc) * DD + kc * 32 + khalf * 16);
      acc = __builtin_amdgcn_wmma_f32_16x16x32_f16(false, A[kc], false, B,
                                                   (short)0, acc, false, false);
    }
    int n = nt * 16 + mloc;
    float bv = bih[n];
    for (int r = 0; r < 8; ++r) {
      int m = khalf * 8 + r;
      pre[(size_t)(row0 + m) * G4 + n] = acc[r] + bv;
    }
  }
}

// ---------------------------------------------------------------------------
// 3) Segment LSTM.  One wave = 16 independent segments; recurrence matmul
//    H(16x128) @ Whh^T(128x512) via WMMA with Whh staged in LDS once per
//    block (the serial recurrence re-reads it every step).  Gate
//    nonlinearities applied on the D-matrix register layout (lane owns
//    N=lane%16, M=8*(lane/16)+r).  Writes final h of non-empty segments.
// ---------------------------------------------------------------------------
__global__ __launch_bounds__(32) void k_lstm(const float* __restrict__ pre,
                                             const int* __restrict__ perm,
                                             const int* __restrict__ seg_start,
                                             const int* __restrict__ seg_len,
                                             const _Float16* __restrict__ Whh,
                                             const float* __restrict__ bhh,
                                             float* __restrict__ emb) {
  __shared__ _Float16 whh_s[G4 * DD];      // 128 KB
  __shared__ float h_s[16][DD];
  __shared__ float c_s[16][DD];
  __shared__ int   edgeIdx[16];
  __shared__ int   lens[16];

  const int lane  = threadIdx.x;
  const int node0 = blockIdx.x * 16;
  const int mloc  = lane & 15;
  const int khalf = lane >> 4;

  stage_to_lds(Whh, whh_s, G4 * DD * 2, lane, 32);

  for (int j = lane; j < 16 * DD; j += 32) {
    (&h_s[0][0])[j] = 0.0f;
    (&c_s[0][0])[j] = 0.0f;
  }
  if (lane < 16) lens[lane] = seg_len[node0 + lane];
  __syncthreads();

  int maxlen = 0;
  for (int i = 0; i < 16; ++i) maxlen = max(maxlen, lens[i]);   // wave-uniform

  for (int t = 0; t < maxlen; ++t) {
    if (lane < 16) {
      int e = -1;
      if (t < lens[lane]) {
        e = perm[seg_start[node0 + lane] + t];
        __builtin_prefetch(pre + (size_t)e * G4, 0, 3);   // global_prefetch_b8
      }
      edgeIdx[lane] = e;
    }
    __syncthreads();

    // A-tiles: current h (16x128) as f16, 4 K-chunks of 32
    v16h A[4];
    for (int kc = 0; kc < 4; ++kc) {
      int kb = kc * 32 + khalf * 16;
      for (int i = 0; i < 16; ++i) A[kc][i] = (_Float16)h_s[mloc][kb + i];
    }

    for (int nt = 0; nt < 8; ++nt) {
      v8f acc[4];
      v8f z = {};
      acc[0] = z; acc[1] = z; acc[2] = z; acc[3] = z;
      for (int kc = 0; kc < 4; ++kc) {
        int kb = kc * 32 + khalf * 16;
        for (int g = 0; g < 4; ++g) {
          int ncol = g * DD + nt * 16 + mloc;       // gate output column
          v16h B = *(const v16h*)(whh_s + ncol * DD + kb);
          acc[g] = __builtin_amdgcn_wmma_f32_16x16x32_f16(false, A[kc], false, B,
                                                          (short)0, acc[g], false, false);
        }
      }
      int j = nt * 16 + mloc;                       // hidden index 0..127
      for (int r = 0; r < 8; ++r) {
        int m = khalf * 8 + r;
        if (t < lens[m]) {
          const float* prow = pre + (size_t)edgeIdx[m] * G4;
          float iv = acc[0][r] + prow[j]          + bhh[j];
          float fv = acc[1][r] + prow[DD + j]     + bhh[DD + j];
          float gv = acc[2][r] + prow[2 * DD + j] + bhh[2 * DD + j];
          float ov = acc[3][r] + prow[3 * DD + j] + bhh[3 * DD + j];
          float si = 1.0f / (1.0f + __expf(-iv));
          float sf = 1.0f / (1.0f + __expf(-fv));
          float so = 1.0f / (1.0f + __expf(-ov));
          float cn = sf * c_s[m][j] + si * tanhf(gv);
          c_s[m][j] = cn;
          h_s[m][j] = so * tanhf(cn);
        }
      }
    }
    __syncthreads();
  }

  for (int m = 0; m < 16; ++m) {
    if (lens[m] > 0) {
      for (int j = lane; j < DD; j += 32)
        emb[(size_t)(node0 + m) * DD + j] = h_s[m][j];
    }
  }
}

// ---------------------------------------------------------------------------
// 4) Head: h1 = [x|emb] @ fc1^T + b; LayerNorm; ReLU; out = h1 @ out_w^T + b
//    One wave per 16 rows; each fc1 tile is consumed exactly once per wave,
//    so no LDS staging (it wouldn't reduce traffic).
// ---------------------------------------------------------------------------
__global__ __launch_bounds__(32) void k_head(const float* __restrict__ x,
                                             const float* __restrict__ emb,
                                             const _Float16* __restrict__ fc1,
                                             const float* __restrict__ fc1_b,
                                             const float* __restrict__ ln_g,
                                             const float* __restrict__ ln_b,
                                             const float* __restrict__ out_w,
                                             const float* __restrict__ out_b,
                                             float* __restrict__ out) {
  __shared__ float h1[16][HID];
  const int lane  = threadIdx.x;
  const int row0  = blockIdx.x * 16;
  const int mloc  = lane & 15;
  const int khalf = lane >> 4;
  const int myrow = row0 + mloc;

  for (int nt = 0; nt < 8; ++nt) {
    v8f acc = {};
    for (int kc = 0; kc < 12; ++kc) {
      int kb = kc * 32 + khalf * 16;
      v16h A;
      for (int i = 0; i < 16; ++i) {
        int k = kb + i;
        float v = (k < FF) ? x[(size_t)myrow * FF + k]
                           : emb[(size_t)myrow * DD + (k - FF)];
        A[i] = (_Float16)v;
      }
      v16h B = *(const v16h*)(fc1 + (size_t)(nt * 16 + mloc) * XC + kb);
      acc = __builtin_amdgcn_wmma_f32_16x16x32_f16(false, A, false, B,
                                                   (short)0, acc, false, false);
    }
    int n = nt * 16 + mloc;
    float bv = fc1_b[n];
    for (int r = 0; r < 8; ++r) {
      int m = khalf * 8 + r;
      h1[m][n] = acc[r] + bv;
    }
  }
  __syncthreads();

  if (lane < 16) {
    int m = lane;
    float mu = 0.0f;
    for (int j = 0; j < HID; ++j) mu += h1[m][j];
    mu *= (1.0f / HID);
    float var = 0.0f;
    for (int j = 0; j < HID; ++j) { float d = h1[m][j] - mu; var += d * d; }
    var *= (1.0f / HID);
    float inv = rsqrtf(var + 1e-5f);
    float o0 = out_b[0], o1 = out_b[1];
    for (int j = 0; j < HID; ++j) {
      float v = (h1[m][j] - mu) * inv * ln_g[j] + ln_b[j];
      v = fmaxf(v, 0.0f);
      o0 += v * out_w[j];
      o1 += v * out_w[HID + j];
    }
    out[(size_t)(row0 + m) * 2 + 0] = o0;
    out[(size_t)(row0 + m) * 2 + 1] = o1;
  }
}

// ---------------------------------------------------------------------------
extern "C" void kernel_launch(void* const* d_in, const int* in_sizes, int n_in,
                              void* d_out, int out_size, void* d_ws, size_t ws_size,
                              hipStream_t stream) {
  const float* x        = (const float*)d_in[0];
  const int*   heads    = (const int*)d_in[1];
  const int*   rels     = (const int*)d_in[2];
  const int*   tails    = (const int*)d_in[3];
  const float* weeks    = (const float*)d_in[4];
  const int*   edge_src = (const int*)d_in[5];   // pre-sorted
  const int*   edge_dst = (const int*)d_in[6];
  const float* ent_embs = (const float*)d_in[10];
  const float* rel_embs = (const float*)d_in[11];
  const float* w_freq   = (const float*)d_in[12];
  const float* w_phi    = (const float*)d_in[13];
  const float* w_amp    = (const float*)d_in[14];
  const float* Wih      = (const float*)d_in[15];
  const float* Whh      = (const float*)d_in[16];
  const float* bih      = (const float*)d_in[17];
  const float* bhh      = (const float*)d_in[18];
  const float* fc1_w    = (const float*)d_in[19];
  const float* fc1_b    = (const float*)d_in[20];
  const float* ln_g     = (const float*)d_in[21];
  const float* ln_b     = (const float*)d_in[22];
  const float* out_w    = (const float*)d_in[23];
  const float* out_b    = (const float*)d_in[24];
  float* out = (float*)d_out;

  // Workspace carve-out (256B aligned slices)
  char* p = (char*)d_ws;
  auto take = [&](size_t bytes) -> char* {
    char* r = p;
    p += (bytes + 255) & ~size_t(255);
    return r;
  };
  _Float16* scoresH   = (_Float16*)take((size_t)N_EDGES * DD * 2);
  float*    pre       = (float*)   take((size_t)N_EDGES * G4 * 4);
  _Float16* WihH      = (_Float16*)take((size_t)G4 * DD * 2);
  _Float16* WhhH      = (_Float16*)take((size_t)G4 * DD * 2);
  _Float16* fc1H      = (_Float16*)take((size_t)HID * XC * 2);
  float*    emb       = (float*)   take((size_t)N_NODES * DD * 4);
  int*      cnts      = (int*)     take((size_t)2 * N_NODES * 4);  // src|dst contiguous
  int*      start_src = (int*)     take((size_t)N_NODES * 4);
  int*      start_dst = (int*)     take((size_t)N_NODES * 4);
  int*      fill      = (int*)     take((size_t)N_NODES * 4);
  int*      perm_s    = (int*)     take((size_t)N_EDGES * 4);
  int*      perm_d    = (int*)     take((size_t)N_EDGES * 4);
  if ((size_t)(p - (char*)d_ws) > ws_size) return;  // not enough scratch
  int* cnt_src = cnts;
  int* cnt_dst = cnts + N_NODES;

  // Weight conversions to f16
  k_cvt_f16<<<(G4 * DD + 255) / 256, 256, 0, stream>>>(Wih, WihH, G4 * DD);
  k_cvt_f16<<<(G4 * DD + 255) / 256, 256, 0, stream>>>(Whh, WhhH, G4 * DD);
  k_cvt_f16<<<(HID * XC + 255) / 256, 256, 0, stream>>>(fc1_w, fc1H, HID * XC);

  // Edge scores
  k_edge_scores<<<(N_EDGES * DD + 255) / 256, 256, 0, stream>>>(
      heads, rels, tails, weeks, ent_embs, rel_embs, w_freq, w_phi, w_amp, scoresH);

  // pre = scores @ Wih^T + bih  (WMMA, Wih staged in LDS per block)
  {
    int tiles = N_EDGES / 16;                // 2500 wave-tiles
    int blocks = (tiles + 7) / 8;            // 8 waves per 256-thread block
    k_pre_gemm<<<blocks, 256, 0, stream>>>(scoresH, WihH, bih, pre);
  }

  // Zero emb + counters, histograms, deterministic stable sort by dst
  k_zero_f32<<<(N_NODES * DD + 255) / 256, 256, 0, stream>>>(emb, N_NODES * DD);
  k_zero_i32<<<(2 * N_NODES + 255) / 256, 256, 0, stream>>>(cnts, 2 * N_NODES);
  k_hist<<<(N_EDGES + 255) / 256, 256, 0, stream>>>(edge_src, cnt_src, N_EDGES);
  k_hist<<<(N_EDGES + 255) / 256, 256, 0, stream>>>(edge_dst, cnt_dst, N_EDGES);
  k_sort_build<<<1, 1, 0, stream>>>(edge_dst, cnt_src, cnt_dst,
                                    start_src, start_dst, perm_s, perm_d, fill);

  // Segment LSTMs (src pass, then dst pass overwrites per reference semantics)
  k_lstm<<<N_NODES / 16, 32, 0, stream>>>(pre, perm_s, start_src, cnt_src, WhhH, bhh, emb);
  k_lstm<<<N_NODES / 16, 32, 0, stream>>>(pre, perm_d, start_dst, cnt_dst, WhhH, bhh, emb);

  // Head MLP + LayerNorm + output (WMMA)
  k_head<<<N_NODES / 16, 32, 0, stream>>>(x, emb, fc1H, fc1_b, ln_g, ln_b,
                                          out_w, out_b, out);
}
